// GaussianHistogram_Angle_17343077941914
// MI455X (gfx1250) — matrix operand order repaired and checked by reading, not compiled, not run on hardware
//
#include <hip/hip_runtime.h>
#include <hip/hip_bf16.h>

typedef __attribute__((ext_vector_type(16))) _Float16 v16h;
typedef __attribute__((ext_vector_type(8)))  float    v8f;

#define BINS   200
#define NCLS   8
#define NTILES 13                   // 13*16 = 208 >= 200 bins
#define DELTA  0.9f                 // (180-0)/200
#define RAD2DEG 57.29577951308232f
#define NEGH_LOG2E  (-0.7213475204444817f)  // -0.5 * log2(e)
#define LOG2_GCONST (-1.4777513f)           // log2(DELTA / sqrt(2*pi))

__global__ __launch_bounds__(256)
void zero_ws_kernel(float* __restrict__ ws) {
    int i = blockIdx.x * blockDim.x + threadIdx.x;
    if (i < NCLS * BINS) ws[i] = 0.0f;
}

// One wave (32 threads) per block. Each wave processes CHUNKS chunks of 32
// angles; the class-segmented Gaussian histogram is accumulated as
// D(16 class x 16 bin) += A(16x32 one-hot, f16) * B(32 angle x 16 bin gauss, f16)
// with 13 bin-tiles kept in f32 WMMA accumulators, flushed once via atomics.
__global__ __launch_bounds__(32)
void angle_hist_wmma(const float* __restrict__ atoms,      // (32,2000,3)
                     const int*   __restrict__ param_idx,  // (32,4000)
                     const int*   __restrict__ atom_idx,   // (32,4000,3)
                     float*       __restrict__ ws)         // (8,200) accum
{
    const int lane = threadIdx.x;   // 0..31
    const int M    = lane & 15;     // A-row (class) / B-col (bin-in-tile)
    const int half = lane >> 4;

    v8f acc[NTILES];
    #pragma unroll
    for (int t = 0; t < NTILES; ++t)
        #pragma unroll
        for (int r = 0; r < 8; ++r)
            acc[t][r] = 0.0f;

    const int CHUNKS = 8;           // 500 blocks * 8 * 32 = 128000 angles
    for (int j = 0; j < CHUNKS; ++j) {
        const int gidx = (blockIdx.x * CHUNKS + j) * 32 + lane;
        const int b = gidx / 4000;
        const int t = gidx - b * 4000;

        const int base3 = (b * 4000 + t) * 3;
        const int i0 = atom_idx[base3 + 0];
        const int i1 = atom_idx[base3 + 1];
        const int i2 = atom_idx[base3 + 2];
        const float* pa = atoms + (size_t)b * 2000 * 3;

        const float p1x = pa[i0*3+0], p1y = pa[i0*3+1], p1z = pa[i0*3+2];
        const float p2x = pa[i1*3+0], p2y = pa[i1*3+1], p2z = pa[i1*3+2];
        const float p3x = pa[i2*3+0], p3y = pa[i2*3+1], p3z = pa[i2*3+2];

        const float v1x = p1x - p2x, v1y = p1y - p2y, v1z = p1z - p2z;
        const float v2x = p3x - p2x, v2y = p3y - p2y, v2z = p3z - p2z;

        const float d11 = v1x*v1x + v1y*v1y + v1z*v1z;
        const float d22 = v2x*v2x + v2y*v2y + v2z*v2z;
        const float d12 = v1x*v2x + v1y*v2y + v1z*v2z;

        float c = d12 / (sqrtf(d11) * sqrtf(d22));
        // NaN-propagating clamp, matching jnp.clip semantics
        c = (c < -0.9999f) ? -0.9999f : c;
        c = (c >  0.9999f) ?  0.9999f : c;
        const float ang = acosf(c) * RAD2DEG;
        const int   cls = param_idx[b * 4000 + t];   // 0..7 == output class row

        // Gather the 16 angles this lane needs for its B slots and build the
        // one-hot A matrix (exact 0/1 in f16).  K order is contiguous, k = 16*half + s, same for A and B.
        float ak[16];
        v16h Am;
        #pragma unroll
        for (int v = 0; v < 8; ++v) {
            const int kB = 16 * half + 2 * v;                 // B: VGPRv <- K,K+1
            ak[2*v]   = __shfl(ang, kB,     32);
            ak[2*v+1] = __shfl(ang, kB + 1, 32);
            const int kA = 16 * half + 2 * v;                 // A: VGPRv <- K,K+1
            const int c0 = __shfl(cls, kA,     32);
            const int c1 = __shfl(cls, kA + 1, 32);
            Am[2*v]   = (c0 == M) ? (_Float16)1.0f : (_Float16)0.0f;
            Am[2*v+1] = (c1 == M) ? (_Float16)1.0f : (_Float16)0.0f;
        }

        #pragma unroll
        for (int tl = 0; tl < NTILES; ++tl) {
            const float center = 0.45f + DELTA * (float)(tl * 16 + M);
            v16h Bv;
            #pragma unroll
            for (int v = 0; v < 16; ++v) {
                // GCONST * exp(-0.5*d^2) == exp2(fma(c1*d, d, c2)):
                // sub, mul, fma, v_exp_f32 — scale folded into the exponent.
                const float d = ak[v] - center;
                const float arg = fmaf(NEGH_LOG2E * d, d, LOG2_GCONST);
                Bv[v] = (_Float16)__builtin_amdgcn_exp2f(arg);
            }
            acc[tl] = __builtin_amdgcn_wmma_f32_16x16x32_f16(
                false, Am, false, Bv, (short)0, acc[tl], false, false);
        }
    }

    // D layout: lanes 0-15 hold N=bin, VGPR r = class r (0..7). Lanes 16-31
    // hold classes 8..15 which are structurally zero -> skip.
    if (half == 0) {
        #pragma unroll
        for (int tl = 0; tl < NTILES; ++tl) {
            const int bin = tl * 16 + lane;
            if (bin < BINS) {
                #pragma unroll
                for (int r = 0; r < 8; ++r)
                    atomicAdd(&ws[r * BINS + bin], acc[tl][r]);
            }
        }
    }
}

// out[bin*8 + cls] = (H[cls,bin]+1e-40) / (sum_b (H[cls,b]+1e-40) + 1e-20)
__global__ __launch_bounds__(256)
void normalize_kernel(const float* __restrict__ ws, float* __restrict__ out) {
    __shared__ float denom[NCLS];
    const int tid = threadIdx.x;
    if (tid < NCLS) {
        float s = 0.0f;
        for (int b = 0; b < BINS; ++b) s += ws[tid * BINS + b] + 1e-40f;
        denom[tid] = s + 1e-20f;
    }
    __syncthreads();
    for (int i = tid; i < BINS * NCLS; i += 256) {
        const int bin = i >> 3;
        const int c   = i & 7;
        out[i] = (ws[c * BINS + bin] + 1e-40f) / denom[c];
    }
}

extern "C" void kernel_launch(void* const* d_in, const int* in_sizes, int n_in,
                              void* d_out, int out_size, void* d_ws, size_t ws_size,
                              hipStream_t stream) {
    const float* atoms     = (const float*)d_in[0];
    const int*   param_idx = (const int*)d_in[1];
    const int*   atom_idx  = (const int*)d_in[2];
    float*       out       = (float*)d_out;
    float*       ws        = (float*)d_ws;

    hipLaunchKernelGGL(zero_ws_kernel, dim3((NCLS*BINS + 255) / 256), dim3(256),
                       0, stream, ws);
    hipLaunchKernelGGL(angle_hist_wmma, dim3(500), dim3(32), 0, stream,
                       atoms, param_idx, atom_idx, ws);
    hipLaunchKernelGGL(normalize_kernel, dim3(1), dim3(256), 0, stream, ws, out);
}
